// SimLoss_66898410602812
// MI455X (gfx1250) — matrix operand order repaired and checked by reading, not compiled, not run on hardware
//
#include <hip/hip_runtime.h>
#include <hip/hip_bf16.h>

typedef __attribute__((ext_vector_type(16))) _Float16 v16h;
typedef __attribute__((ext_vector_type(8)))  _Float16 v8h;
typedef __attribute__((ext_vector_type(8)))  float    v8f;

#define BN 8192   // batch
#define DK 128    // feature dim

// ---------------------------------------------------------------------------
// Kernel 1: row L2-normalize fp32 -> f16, plus self-dot of the *rounded* f16
// row (SD[i] = sum (f16(x_i/n_i))^2 in f32 == the WMMA diagonal S_ii).
// One 128-thread block per row.
// ---------------------------------------------------------------------------
__global__ void __launch_bounds__(128) norm_rows(const float* __restrict__ X,
                                                 _Float16* __restrict__ Xh,
                                                 float* __restrict__ SD) {
  const int row = blockIdx.x;
  const int t   = threadIdx.x;            // 0..127
  const float x = X[(size_t)row * DK + t];
  float ss = x * x;
  #pragma unroll
  for (int m = 16; m >= 1; m >>= 1) ss += __shfl_xor(ss, m, 32);
  __shared__ float lds[4];
  __shared__ float rinv_s;
  const int wave = t >> 5;
  if ((t & 31) == 0) lds[wave] = ss;
  __syncthreads();
  if (t == 0) {
    float tot = lds[0] + lds[1] + lds[2] + lds[3];
    rinv_s = 1.0f / fmaxf(sqrtf(tot), 1e-12f);
  }
  __syncthreads();
  const _Float16 h = (_Float16)(x * rinv_s);
  Xh[(size_t)row * DK + t] = h;

  // self-dot of the f16-rounded normalized row (for diagonal removal)
  const float hf = (float)h;
  float sd = hf * hf;
  #pragma unroll
  for (int m = 16; m >= 1; m >>= 1) sd += __shfl_xor(sd, m, 32);
  __syncthreads();
  if ((t & 31) == 0) lds[wave] = sd;
  __syncthreads();
  if (t == 0) SD[row] = lds[0] + lds[1] + lds[2] + lds[3];
}

// ---------------------------------------------------------------------------
// Kernel 2: fused S = Xh*Xh^T  ->  Z[i] = sum_{j != i} exp(S[i,j]).
// 256 WGs x 256 threads (8 waves).  WG owns 32 rows (A-frags live in regs).
// Wave w covers column tiles j0 = it*128 + w*16, it = 0..63.  B tiles are
// double-buffered; diagonal term is NOT masked in the loop — it is
// subtracted once per row at the end via exp(SD[i]).
// ---------------------------------------------------------------------------
__global__ void __launch_bounds__(256, 1) cos_z(const _Float16* __restrict__ Xh,
                                                const float* __restrict__ SD,
                                                float* __restrict__ Z) {
  const int i0   = blockIdx.x * 32;
  const int lane = threadIdx.x & 31;
  const int wave = threadIdx.x >> 5;
  const int hs   = (lane >> 4) & 1;       // half-wave select
  const int l16  = lane & 15;

  // A fragments for 32 rows x 128 K (64 VGPRs), reused for all 64 col tiles.
  // A 16x32 k order used here (contiguous, not the ISA table order): lane(0-15): a[0..15]=K0..15;
  // lane(16-31): a[0..15]=K16..31 (per K-step of 32).
  v16h a[2][4];
  #pragma unroll
  for (int h = 0; h < 2; ++h) {
    const _Float16* base = Xh + (size_t)(i0 + h * 16 + l16) * DK + hs * 16;
    #pragma unroll
    for (int ks = 0; ks < 4; ++ks) {
      v8h lo = *(const v8h*)(base + ks * 32);
      v8h hi = *(const v8h*)(base + ks * 32 + 8);
      #pragma unroll
      for (int t = 0; t < 8; ++t) { a[h][ks][t] = lo[t]; a[h][ks][t + 8] = hi[t]; }
    }
  }

  float zp[2][8];
  #pragma unroll
  for (int h = 0; h < 2; ++h)
    #pragma unroll
    for (int r = 0; r < 8; ++r) zp[h][r] = 0.0f;

  // B 32x16 k order used here (contiguous, same as A): N = lane%16; lanes 0-15 -> K0..15,
  // lanes 16-31 -> K16..31 => 16 contiguous halves per lane per K-step.
  auto loadB = [&](v16h* dst, int it) {
    const int j0 = it * 128 + wave * 16;
    const _Float16* bb = Xh + (size_t)(j0 + l16) * DK + hs * 16;
    #pragma unroll
    for (int ks = 0; ks < 4; ++ks) dst[ks] = *(const v16h*)(bb + ks * 32);
  };

  v16h bc[4], bn[4];
  loadB(bc, 0);

  #pragma unroll 2
  for (int it = 0; it < 64; ++it) {
    if (it + 1 < 64) loadB(bn, it + 1);                    // next tile in flight
    if (it + 2 < 64)                                       // warm WGP cache
      __builtin_prefetch(Xh + (size_t)((it + 2) * 128 + wave * 16 + l16) * DK, 0, 3);

    #pragma unroll
    for (int h = 0; h < 2; ++h) {
      v8f c = {};
      #pragma unroll
      for (int ks = 0; ks < 4; ++ks)
        c = __builtin_amdgcn_wmma_f32_16x16x32_f16(
                false, a[h][ks], false, bc[ks], (short)0, c, false, false);
      #pragma unroll
      for (int r = 0; r < 8; ++r) zp[h][r] += __expf(c[r]);  // diag included
    }
    #pragma unroll
    for (int q = 0; q < 4; ++q) bc[q] = bn[q];             // renamed away by unroll
  }

  // Reduce each row partial across the 16 lanes of its half-wave group.
  #pragma unroll
  for (int h = 0; h < 2; ++h)
    #pragma unroll
    for (int r = 0; r < 8; ++r) {
      float v = zp[h][r];
      v += __shfl_xor(v, 1, 32);
      v += __shfl_xor(v, 2, 32);
      v += __shfl_xor(v, 4, 32);
      v += __shfl_xor(v, 8, 32);
      zp[h][r] = v;
    }

  // Cross-wave reduce in LDS, fixed summation order (deterministic).
  __shared__ float lds[8][32];
  if (l16 == 0) {                  // lanes 0 and 16 hold the reduced values
    #pragma unroll
    for (int h = 0; h < 2; ++h)
      #pragma unroll
      for (int r = 0; r < 8; ++r) lds[wave][h * 16 + hs * 8 + r] = zp[h][r];
  }
  __syncthreads();
  if (threadIdx.x < 32) {
    float s = 0.0f;
    #pragma unroll
    for (int w = 0; w < 8; ++w) s += lds[w][threadIdx.x];
    const int row = i0 + threadIdx.x;
    Z[row] = s - __expf(SD[row]);          // remove diagonal exp(S_ii)
  }
}

// ---------------------------------------------------------------------------
// Kernel 3: pair terms + final scalar.  pairs (2k, 2k+1), k = 0..4094.
// loss_k = -2*dot(xh_2k, xh_2k+1) + log(Z[2k]) + log(Z[2k+1]); out = sum/BN.
// ---------------------------------------------------------------------------
__global__ void __launch_bounds__(256) pair_loss(const _Float16* __restrict__ Xh,
                                                 const float* __restrict__ Z,
                                                 float* __restrict__ out) {
  const int npairs = (BN + 1) / 2 - 1;   // 4095
  float acc = 0.0f;
  for (int k = threadIdx.x; k < npairs; k += blockDim.x) {
    const _Float16* xa = Xh + (size_t)(2 * k) * DK;
    const _Float16* xb = xa + DK;
    float d = 0.0f;
    #pragma unroll 8
    for (int t = 0; t < DK; ++t) d += (float)xa[t] * (float)xb[t];
    acc += -2.0f * d + __logf(Z[2 * k]) + __logf(Z[2 * k + 1]);
  }
  #pragma unroll
  for (int m = 16; m >= 1; m >>= 1) acc += __shfl_xor(acc, m, 32);
  __shared__ float lds[8];
  const int wave = threadIdx.x >> 5;
  if ((threadIdx.x & 31) == 0) lds[wave] = acc;
  __syncthreads();
  if (threadIdx.x == 0) {
    float s = 0.0f;
    #pragma unroll
    for (int w = 0; w < 8; ++w) s += lds[w];
    out[0] = s / (float)BN;
  }
}

// ---------------------------------------------------------------------------
extern "C" void kernel_launch(void* const* d_in, const int* in_sizes, int n_in,
                              void* d_out, int out_size, void* d_ws, size_t ws_size,
                              hipStream_t stream) {
  const float* X  = (const float*)d_in[0];
  char*        ws = (char*)d_ws;
  _Float16*    Xh = (_Float16*)ws;                               // 2 MB
  float*       Z  = (float*)(ws + (size_t)BN * DK * 2);          // 32 KB
  float*       SD = (float*)(ws + (size_t)BN * DK * 2 + BN * 4); // 32 KB
  float*       out = (float*)d_out;

  norm_rows<<<BN, 128, 0, stream>>>(X, Xh, SD);
  cos_z<<<BN / 32, 256, 0, stream>>>(Xh, SD, Z);
  pair_loss<<<1, 256, 0, stream>>>(Xh, Z, out);
}
